// SimpleFineGrainedHeadWeight_87591563034911
// MI455X (gfx1250) — matrix operand order repaired and checked by reading, not compiled, not run on hardware
//
#include <hip/hip_runtime.h>
#include <math.h>

typedef __attribute__((ext_vector_type(16))) _Float16 v16h;
typedef __attribute__((ext_vector_type(8)))  _Float16 v8h;
typedef __attribute__((ext_vector_type(8)))  float    v8f;
typedef __attribute__((ext_vector_type(2)))  float    v2f;

#define B_   64
#define V_   196
#define T_   64
#define VIS_ 768
#define E_   512
#define MV_  (B_ * V_)   /* 12544 */
#define MT_  (B_ * T_)   /* 4096  */

// ---------------------------------------------------------------------------
// WMMA helpers (CDNA5 gfx1250, wave32).
// ---------------------------------------------------------------------------
static __device__ __forceinline__ v8f wmma16(v16h a, v16h b, v8f c) {
  return __builtin_amdgcn_wmma_f32_16x16x32_f16(
      /*neg_a=*/false, a, /*neg_b=*/false, b,
      /*c_mod=*/(short)0, c, /*reuse_a=*/false, /*reuse_b=*/false);
}

// Full-precision f32 WMMA: D(16x16) = A(16x4) * B(4x16) + C.
// A: lane half hi holds K = 2*hi + {0,1} for row M = lane%16; B mirrors on N.
static __device__ __forceinline__ v8f wmma4(v2f a, v2f b, v8f c) {
  return __builtin_amdgcn_wmma_f32_16x16x4_f32(
      /*neg_a=*/false, a, /*neg_b=*/false, b,
      /*c_mod=*/(short)0, c, /*reuse_a=*/false, /*reuse_b=*/false);
}

// A-matrix 16x32 f16 fragment from a row-major f16 row pointer (global or LDS).
// ISA layout: lane half hi=0 holds K = kk+[0..7] and kk+16+[0..7];
//             lane half hi=1 holds K = kk+8+[0..7] and kk+24+[0..7].
static __device__ __forceinline__ v16h a_frag_f16(const _Float16* __restrict__ row,
                                                  int kk, int hi) {
  const int k0 = kk + hi * 8;
  const int k1 = kk + 16 + hi * 8;
  v8h lo = *reinterpret_cast<const v8h*>(row + k0);
  v8h h2 = *reinterpret_cast<const v8h*>(row + k1);
  return __builtin_shufflevector(lo, h2, 0, 1, 2, 3, 4, 5, 6, 7,
                                 8, 9, 10, 11, 12, 13, 14, 15);
}

// B-matrix 32x16 f16 fragment. Lane n<16 holds column n, K = kk+[0..7] and kk+16+[0..7];
// lane n+16 holds column n, K = kk+8+[0..7] and kk+24+[0..7] (matches A fragment).
static __device__ __forceinline__ v16h b_frag_f16(const _Float16* __restrict__ colrow,
                                                  int kk, int hi) {
  const int k0 = kk + hi * 8;
  v8h lo = *reinterpret_cast<const v8h*>(colrow + k0);
  v8h h2 = *reinterpret_cast<const v8h*>(colrow + k0 + 16);
  return __builtin_shufflevector(lo, h2, 0, 1, 2, 3, 4, 5, 6, 7,
                                 8, 9, 10, 11, 12, 13, 14, 15);
}

// ---------------------------------------------------------------------------
// Pack W [768x512] f32 -> WT [512x768] f16 (transpose for row-wise B frags)
// ---------------------------------------------------------------------------
__global__ __launch_bounds__(256) void packW_kernel(const float* __restrict__ W,
                                                    _Float16* __restrict__ WT) {
  int i = blockIdx.x * 256 + threadIdx.x;
  if (i < E_ * VIS_) {
    int e = i / VIS_, k = i % VIS_;
    WT[i] = (_Float16)W[(size_t)k * E_ + e];
  }
}

// ---------------------------------------------------------------------------
// Projection: out[M x 512] = X[M x 768] @ W + bias.
// Block = 64 rows x 512 cols, 256 threads (8 waves); wave w owns cols
// [w*64, w*64+64) with a 4(m-strip) x 4(n-tile) register accumulator.
// A strip converted f32->f16 into LDS (two 384-wide K stages), shared by all
// waves; B frags stream from the f16 WT in L2.
// ---------------------------------------------------------------------------
#define PROJ_LD 400  /* staged halfs per row: 384 data + 16 pad (16B align) */
__global__ __launch_bounds__(256) void proj_kernel(const float* __restrict__ X,
                                                   const _Float16* __restrict__ WT,
                                                   const float* __restrict__ bias,
                                                   float* __restrict__ out) {
  const int blk = blockIdx.x;
  const int lane = threadIdx.x & 31;
  const int wave = threadIdx.x >> 5;
  const int hl = lane & 15, hi = lane >> 4;
  __shared__ __attribute__((aligned(16))) _Float16 Xs[64 * PROJ_LD];  // 51.2 KB

  v8f acc[4][4];
#pragma unroll
  for (int ms = 0; ms < 4; ++ms)
#pragma unroll
    for (int nt = 0; nt < 4; ++nt)
#pragma unroll
      for (int j = 0; j < 8; ++j) acc[ms][nt][j] = 0.f;

  for (int half = 0; half < 2; ++half) {
    const int kbase = half * 384;
    __syncthreads();
    for (int c = threadIdx.x; c < 3072; c += 256) {
      const int row = c / 48;
      const int k8 = (c % 48) * 8;
      const float* src = X + ((size_t)blk * 64 + row) * VIS_ + kbase + k8;
      float4 f0 = *reinterpret_cast<const float4*>(src);
      float4 f1 = *reinterpret_cast<const float4*>(src + 4);
      v8h h;
      h[0] = (_Float16)f0.x; h[1] = (_Float16)f0.y;
      h[2] = (_Float16)f0.z; h[3] = (_Float16)f0.w;
      h[4] = (_Float16)f1.x; h[5] = (_Float16)f1.y;
      h[6] = (_Float16)f1.z; h[7] = (_Float16)f1.w;
      *reinterpret_cast<v8h*>(Xs + row * PROJ_LD + k8) = h;
    }
    __syncthreads();
    for (int kk = 0; kk < 384; kk += 32) {
      v16h bb[4];
#pragma unroll
      for (int nt = 0; nt < 4; ++nt)
        bb[nt] = b_frag_f16(WT + (size_t)(wave * 64 + nt * 16 + hl) * VIS_ + kbase,
                            kk, hi);
#pragma unroll
      for (int ms = 0; ms < 4; ++ms) {
        v16h a = a_frag_f16(Xs + (ms * 16 + hl) * PROJ_LD, kk, hi);
#pragma unroll
        for (int nt = 0; nt < 4; ++nt) acc[ms][nt] = wmma16(a, bb[nt], acc[ms][nt]);
      }
    }
  }
#pragma unroll
  for (int ms = 0; ms < 4; ++ms)
#pragma unroll
    for (int nt = 0; nt < 4; ++nt)
#pragma unroll
      for (int j = 0; j < 8; ++j) {
        const int row = blk * 64 + ms * 16 + j + hi * 8;
        const int col = wave * 64 + nt * 16 + hl;
        out[(size_t)row * E_ + col] = acc[ms][nt][j] + bias[col];
      }
}

// ---------------------------------------------------------------------------
// Per-row: wraw = dot(row, wfc)+bfc (PRE-norm), then l2-normalize row in place
// and emit f16 copy. grid = nrows, block = 128.
// ---------------------------------------------------------------------------
__global__ __launch_bounds__(128) void rownorm_kernel(float* __restrict__ mat,
                                                      _Float16* __restrict__ mat16,
                                                      const float* __restrict__ wfc,
                                                      const float* __restrict__ bfc,
                                                      float* __restrict__ wraw) {
  const int row = blockIdx.x;
  const int t = threadIdx.x;
  float* p = mat + (size_t)row * E_;
  float v[4];
  float ss = 0.f, dot = 0.f;
#pragma unroll
  for (int i = 0; i < 4; ++i) {
    v[i] = p[t + i * 128];
    ss += v[i] * v[i];
    dot += v[i] * wfc[t + i * 128];
  }
  __shared__ float s1[128], s2[128];
  s1[t] = ss; s2[t] = dot;
  __syncthreads();
  for (int o = 64; o > 0; o >>= 1) {
    if (t < o) { s1[t] += s1[t + o]; s2[t] += s2[t + o]; }
    __syncthreads();
  }
  const float inv = 1.f / fmaxf(sqrtf(s1[0]), 1e-12f);
  if (t == 0) wraw[row] = s2[0] + bfc[0];
#pragma unroll
  for (int i = 0; i < 4; ++i) {
    float nv = v[i] * inv;
    p[t + i * 128] = nv;
    mat16[(size_t)row * E_ + t + i * 128] = (_Float16)nv;
  }
}

// ---------------------------------------------------------------------------
// Softmax over V=196 per batch (no mask). In-place on vw[B,V].
// ---------------------------------------------------------------------------
__global__ __launch_bounds__(256) void softmax_v_kernel(float* __restrict__ vw) {
  const int b = blockIdx.x, t = threadIdx.x;
  __shared__ float red[256];
  float x = (t < V_) ? vw[b * V_ + t] : -INFINITY;
  red[t] = x;
  __syncthreads();
  for (int o = 128; o > 0; o >>= 1) {
    if (t < o) red[t] = fmaxf(red[t], red[t + o]);
    __syncthreads();
  }
  const float m = red[0];
  __syncthreads();
  float e = (t < V_) ? expf(x - m) : 0.f;
  red[t] = e;
  __syncthreads();
  for (int o = 128; o > 0; o >>= 1) {
    if (t < o) red[t] += red[t + o];
    __syncthreads();
  }
  if (t < V_) vw[b * V_ + t] = e / red[0];
}

// ---------------------------------------------------------------------------
// Masked softmax over T=64 per batch (pos >= len -> prob 0). In-place tw[B,T].
// ---------------------------------------------------------------------------
__global__ __launch_bounds__(64) void softmax_t_kernel(float* __restrict__ tw,
                                                       const int* __restrict__ tlen) {
  const int b = blockIdx.x, t = threadIdx.x;
  __shared__ float red[64];
  const int len = tlen[b];
  float x = (t < len) ? tw[b * T_ + t] : -INFINITY;
  red[t] = x;
  __syncthreads();
  for (int o = 32; o > 0; o >>= 1) {
    if (t < o) red[t] = fmaxf(red[t], red[t + o]);
    __syncthreads();
  }
  const float m = red[0];
  __syncthreads();
  float e = (t < len) ? expf(x - m) : 0.f;
  red[t] = e;
  __syncthreads();
  for (int o = 32; o > 0; o >>= 1) {
    if (t < o) red[t] += red[t + o];
    __syncthreads();
  }
  tw[b * T_ + t] = (t < len) ? e / red[0] : 0.f;
}

// ---------------------------------------------------------------------------
// Fused cross-batch similarity: grid = (b, q-pair), block = 416 (13 waves).
// K is processed in 8 windows of 64, double-buffered in LDS via the gfx1250
// async Global->LDS path: waves 0..7 each issue exactly 4 b128 transfers per
// stage; stage s+1 is issued before `s_wait_asynccnt 4`, so (in-order ASYNCcnt
// retirement) stage s is resident while s+1 streams behind the WMMAs.
// The 52 KB C tile overlays both staging buffers for the reductions.
// ---------------------------------------------------------------------------
#define KW 64                 /* staged K window (halfs) */
#define STAGE_HALFS (2 * 64 * KW)  /* 8192 halfs = 16 KB per buffer */

static __device__ __forceinline__ void sim_stage_issue(
    const _Float16* __restrict__ te16, int q0, int stage, unsigned ldsbase,
    int tid) {
  if (tid < 256) {
    const int kbase = stage * KW;
#pragma unroll
    for (int i = 0; i < 4; ++i) {
      const int c = tid + i * 256;       // 1024 chunks: [qi:1][row:6][k8:3]
      const int qi = c >> 9;
      const int r = (c >> 3) & 63;
      const int k8 = (c & 7) * 8;
      const uint64_t g = (uint64_t)(uintptr_t)(
          te16 + ((size_t)(q0 + qi) * T_ + r) * E_ + kbase + k8);
      const unsigned l = ldsbase +
          (unsigned)((((stage & 1) * STAGE_HALFS) + (qi * 64 + r) * KW + k8) * 2);
      asm volatile("global_load_async_to_lds_b128 %0, %1, off"
                   :: "v"(l), "v"(g) : "memory");
    }
  }
}

__global__ __launch_bounds__(416) void sim_kernel(const _Float16* __restrict__ ve16,
                                                  const _Float16* __restrict__ te16,
                                                  const float* __restrict__ vw,
                                                  const float* __restrict__ tw,
                                                  float* __restrict__ out_i2t,
                                                  float* __restrict__ out_t2i,
                                                  int* __restrict__ idx1,
                                                  int* __restrict__ idx2) {
  const int b = blockIdx.x, q0 = blockIdx.y * 2;
  const int tid = threadIdx.x;
  const int lane = tid & 31, wave = tid >> 5;  // wave 0..12
  const int hl = lane & 15, hi = lane >> 4;

  // 53248 B: double-buffered te stage (2 x 16 KB) overlaid by Cs[208][64] f32.
  __shared__ __attribute__((aligned(16))) char smem[208 * 64 * 4];
  __shared__ float redV[208];
  __shared__ float redT[64];
  _Float16* teS = (_Float16*)smem;
  float* Cs = (float*)smem;
  const unsigned ldsbase = (unsigned)(uintptr_t)teS;

  const int vrow = wave * 16 + hl;
  const bool avalid = (vrow < V_);
  const _Float16* arow = ve16 + ((size_t)b * V_ + (avalid ? vrow : 0)) * E_;

  v8f acc[2][4];
#pragma unroll
  for (int qi = 0; qi < 2; ++qi)
#pragma unroll
    for (int nt = 0; nt < 4; ++nt)
#pragma unroll
      for (int j = 0; j < 8; ++j) acc[qi][nt][j] = 0.f;

  sim_stage_issue(te16, q0, 0, ldsbase, tid);
  for (int s = 0; s < 8; ++s) {
    if (s < 7) {
      sim_stage_issue(te16, q0, s + 1, ldsbase, tid);
      asm volatile("s_wait_asynccnt 0x4" ::: "memory");
    } else {
      asm volatile("s_wait_asynccnt 0x0" ::: "memory");
    }
    __syncthreads();  // stage s visible to all waves
    const int kbase = s * KW;
    const _Float16* bufp = teS + (s & 1) * STAGE_HALFS;
    for (int kk = 0; kk < KW; kk += 32) {
      v16h a = a_frag_f16(arow + kbase, kk, hi);
      if (!avalid) {
#pragma unroll
        for (int i = 0; i < 16; ++i) a[i] = (_Float16)0.f;
      }
#pragma unroll
      for (int qi = 0; qi < 2; ++qi)
#pragma unroll
        for (int nt = 0; nt < 4; ++nt) {
          v16h bb = b_frag_f16(bufp + (size_t)(qi * 64 + nt * 16 + hl) * KW, kk, hi);
          acc[qi][nt] = wmma16(a, bb, acc[qi][nt]);
        }
    }
    __syncthreads();  // buffer reusable two stages later
  }

  for (int qi = 0; qi < 2; ++qi) {
    const int q = q0 + qi;
    __syncthreads();
#pragma unroll
    for (int nt = 0; nt < 4; ++nt)
#pragma unroll
      for (int j = 0; j < 8; ++j)
        Cs[(wave * 16 + j + hi * 8) * 64 + nt * 16 + hl] = acc[qi][nt][j];
    __syncthreads();

    if (tid < V_) {  // per visual token: max over text tokens
      float m = -INFINITY;
      int arg = 0;
      for (int t = 0; t < T_; ++t) {
        float c = Cs[tid * 64 + t];
        if (c > m) { m = c; arg = t; }
      }
      redV[tid] = vw[b * V_ + tid] * m;
      if (b == q) idx1[b * V_ + tid] = arg;
    }
    if (tid >= 256 && tid < 256 + T_) {  // per text token: max over visual tokens
      const int t = tid - 256;
      float m = -INFINITY;
      int arg = 0;
      for (int v = 0; v < V_; ++v) {
        float c = Cs[v * 64 + t];
        if (c > m) { m = c; arg = v; }
      }
      redT[t] = tw[q * T_ + t] * m;
      if (b == q) idx2[q * T_ + t] = arg;
    }
    __syncthreads();
    if (tid == 0) {
      float s = 0.f;
      for (int v = 0; v < V_; ++v) s += redV[v];
      out_i2t[b * B_ + q] = s;
    }
    if (tid == 1) {
      float s = 0.f;
      for (int t = 0; t < T_; ++t) s += redT[t];
      out_t2i[b * B_ + q] = s;
    }
  }
}

// ---------------------------------------------------------------------------
// Column mean / unbiased inv-std for the 4 zn() matrices.
// blockIdx.x = matrix id {0:t_feat, 1:max_v, 2:v_feat, 3:max_t}; thread = col.
// ---------------------------------------------------------------------------
__global__ __launch_bounds__(512) void stats_kernel(const float* __restrict__ ve_n,
                                                    const float* __restrict__ te_n,
                                                    const int* __restrict__ idx1,
                                                    const int* __restrict__ idx2,
                                                    float* __restrict__ stats) {
  const int mat = blockIdx.x;
  const int col = threadIdx.x;
  const int n = (mat < 2) ? MT_ : MV_;
  float s = 0.f, sq = 0.f;
  for (int a = 0; a < n; ++a) {
    float x;
    if (mat == 0) {
      x = te_n[(size_t)a * E_ + col];
    } else if (mat == 1) {
      const int bq = a >> 6;
      x = ve_n[((size_t)bq * V_ + idx2[a]) * E_ + col];
    } else if (mat == 2) {
      x = ve_n[(size_t)a * E_ + col];
    } else {
      const int bq = a / V_;
      x = te_n[((size_t)bq * T_ + idx1[a]) * E_ + col];
    }
    s += x;
    sq += x * x;
  }
  const float mean = s / (float)n;
  const float var = (sq - (float)n * mean * mean) / (float)(n - 1);
  stats[(mat * 2 + 0) * E_ + col] = mean;
  stats[(mat * 2 + 1) * E_ + col] = rsqrtf(fmaxf(var, 1e-24f));
}

// ---------------------------------------------------------------------------
// c = (c1 + c2)/2 via V_WMMA_F32_16X16X4_F32 (full f32 precision).
// Block = 256 (8 waves); each wave owns one 16x16 output tile.
// Per 4-row step: lane half hi supplies rows a0+2*hi+{0,1}; A = zn(left) col
// (cbase+hl), B = zn(right) col (dbase+hl) scaled by the per-row weight.
// ---------------------------------------------------------------------------
__global__ __launch_bounds__(256) void cmat_kernel(const float* __restrict__ ve_n,
                                                   const float* __restrict__ te_n,
                                                   const int* __restrict__ idx1,
                                                   const int* __restrict__ idx2,
                                                   const float* __restrict__ vw,
                                                   const float* __restrict__ tw,
                                                   const float* __restrict__ stats,
                                                   float* __restrict__ cmat) {
  const int lane = threadIdx.x & 31, wave = threadIdx.x >> 5;
  const int hl = lane & 15, hi = lane >> 4;
  const int cbase = blockIdx.y * 16;
  const int dbase = (blockIdx.x * 8 + wave) * 16;
  const int ccol = cbase + hl, dcol = dbase + hl;

  v8f acc;
#pragma unroll
  for (int j = 0; j < 8; ++j) acc[j] = 0.f;

#pragma unroll 1
  for (int phase = 0; phase < 2; ++phase) {
    const int n = (phase == 0) ? MT_ : MV_;
    const float muL = stats[(phase ? 4 : 0) * E_ + ccol];
    const float isL = stats[(phase ? 5 : 1) * E_ + ccol];
    const float muR = stats[(phase ? 6 : 2) * E_ + dcol];
    const float isR = stats[(phase ? 7 : 3) * E_ + dcol];
    for (int a0 = 0; a0 < n; a0 += 4) {
      const int a = a0 + hi * 2;
      float l0, l1, r0, r1, w0, w1;
      if (phase == 0) {
        l0 = te_n[(size_t)a * E_ + ccol];
        l1 = te_n[(size_t)(a + 1) * E_ + ccol];
        r0 = ve_n[((size_t)(a >> 6) * V_ + idx2[a]) * E_ + dcol];
        r1 = ve_n[((size_t)((a + 1) >> 6) * V_ + idx2[a + 1]) * E_ + dcol];
        w0 = tw[a]; w1 = tw[a + 1];
      } else {
        l0 = ve_n[(size_t)a * E_ + ccol];
        l1 = ve_n[(size_t)(a + 1) * E_ + ccol];
        r0 = te_n[((size_t)(a / V_) * T_ + idx1[a]) * E_ + dcol];
        r1 = te_n[((size_t)((a + 1) / V_) * T_ + idx1[a + 1]) * E_ + dcol];
        w0 = vw[a]; w1 = vw[a + 1];
      }
      v2f av, bv;
      av[0] = (l0 - muL) * isL;
      av[1] = (l1 - muL) * isL;
      bv[0] = (r0 - muR) * isR * w0;
      bv[1] = (r1 - muR) * isR * w1;
      acc = wmma4(av, bv, acc);
    }
  }
#pragma unroll
  for (int j = 0; j < 8; ++j)
    cmat[(size_t)(cbase + j + hi * 8) * E_ + dcol] = acc[j] * (0.5f / (float)B_);
}

// ---------------------------------------------------------------------------
// cdcr loss: 0.1 * (sum_i (c_ii-1)^2 + 0.06 * sum_{i!=j} c_ij^2).
// ---------------------------------------------------------------------------
__global__ __launch_bounds__(1024) void loss_kernel(const float* __restrict__ cmat,
                                                    float* __restrict__ out_loss) {
  const int t = threadIdx.x;
  float on = 0.f, off = 0.f;
  for (int i = t; i < E_ * E_; i += 1024) {
    const float v = cmat[i];
    const int r = i >> 9, c = i & 511;
    if (r == c) { const float d = v - 1.f; on += d * d; }
    else        { off += v * v; }
  }
  __shared__ float s1[1024], s2[1024];
  s1[t] = on; s2[t] = off;
  __syncthreads();
  for (int o = 512; o > 0; o >>= 1) {
    if (t < o) { s1[t] += s1[t + o]; s2[t] += s2[t + o]; }
    __syncthreads();
  }
  if (t == 0) out_loss[0] = 0.1f * (s1[0] + 0.06f * s2[0]);
}

// ---------------------------------------------------------------------------
extern "C" void kernel_launch(void* const* d_in, const int* in_sizes, int n_in,
                              void* d_out, int out_size, void* d_ws, size_t ws_size,
                              hipStream_t stream) {
  (void)in_sizes; (void)n_in; (void)out_size; (void)ws_size;
  const float* v_tok  = (const float*)d_in[1];
  const float* t_tok  = (const float*)d_in[3];
  const float* Wv_tok = (const float*)d_in[8];
  const float* bv_tok = (const float*)d_in[9];
  const float* Wt_tok = (const float*)d_in[10];
  const float* bt_tok = (const float*)d_in[11];
  const float* wv_fc  = (const float*)d_in[12];
  const float* bv_fc  = (const float*)d_in[13];
  const float* wt_fc  = (const float*)d_in[14];
  const float* bt_fc  = (const float*)d_in[15];
  const int*   tlen   = (const int*)d_in[16];

  char* ws = (char*)d_ws;
  size_t off = 0;
  auto carve = [&](size_t bytes) -> void* {
    off = (off + 255) & ~(size_t)255;
    void* p = ws + off;
    off += bytes;
    return p;
  };
  _Float16* WvT  = (_Float16*)carve((size_t)E_ * VIS_ * 2);
  _Float16* WtT  = (_Float16*)carve((size_t)E_ * VIS_ * 2);
  float*    ve   = (float*)carve((size_t)MV_ * E_ * 4);
  float*    te   = (float*)carve((size_t)MT_ * E_ * 4);
  _Float16* ve16 = (_Float16*)carve((size_t)MV_ * E_ * 2);
  _Float16* te16 = (_Float16*)carve((size_t)MT_ * E_ * 2);
  float*    vw   = (float*)carve((size_t)MV_ * 4);
  float*    tw   = (float*)carve((size_t)MT_ * 4);
  int*      idx1 = (int*)carve((size_t)MV_ * 4);
  int*      idx2 = (int*)carve((size_t)MT_ * 4);
  float*    stats = (float*)carve((size_t)8 * E_ * 4);
  float*    cmat  = (float*)carve((size_t)E_ * E_ * 4);

  float* out_i2t  = (float*)d_out;          // [B,B]
  float* out_t2i  = out_i2t + B_ * B_;      // [B,B]
  float* out_loss = out_i2t + 2 * B_ * B_;  // [1]

  packW_kernel<<<(E_ * VIS_ + 255) / 256, 256, 0, stream>>>(Wv_tok, WvT);
  packW_kernel<<<(E_ * VIS_ + 255) / 256, 256, 0, stream>>>(Wt_tok, WtT);

  proj_kernel<<<MV_ / 64, 256, 0, stream>>>(v_tok, WvT, bv_tok, ve);
  proj_kernel<<<MT_ / 64, 256, 0, stream>>>(t_tok, WtT, bt_tok, te);

  rownorm_kernel<<<MV_, 128, 0, stream>>>(ve, ve16, wv_fc, bv_fc, vw);
  rownorm_kernel<<<MT_, 128, 0, stream>>>(te, te16, wt_fc, bt_fc, tw);

  softmax_v_kernel<<<B_, 256, 0, stream>>>(vw);
  softmax_t_kernel<<<B_, 64, 0, stream>>>(tw, tlen);

  sim_kernel<<<dim3(B_, B_ / 2), 416, 0, stream>>>(ve16, te16, vw, tw,
                                                   out_i2t, out_t2i, idx1, idx2);

  stats_kernel<<<4, 512, 0, stream>>>(ve, te, idx1, idx2, stats);
  cmat_kernel<<<dim3(4, 32), 256, 0, stream>>>(ve, te, idx1, idx2,
                                               vw, tw, stats, cmat);
  loss_kernel<<<1, 1024, 0, stream>>>(cmat, out_loss);
}
